// ModuleAttentionProp_74947179315253
// MI455X (gfx1250) — matrix operand order repaired and checked by reading, not compiled, not run on hardware
//
#include <hip/hip_runtime.h>
#include <cstdint>
#include <cstddef>

typedef __bf16 bf16;
typedef __attribute__((ext_vector_type(16))) __bf16 v16bf;
typedef __attribute__((ext_vector_type(8))) __bf16 v8bf;
typedef __attribute__((ext_vector_type(8))) float v8f;

#define Bc 2
#define Kc 256
#define Nc 2048
#define Dc 1024
#define Hc 128
#define NBc 10
#define DEc 64

// ---------------------------------------------------------------------------
// CDNA5 async global->LDS helpers (inline asm: CDNA5 ISA)
// ---------------------------------------------------------------------------
static __device__ __forceinline__ uint32_t lds_lo32(const void* p) {
  return (uint32_t)(uintptr_t)p;  // low 32 bits of generic LDS pointer
}
static __device__ __forceinline__ void async_ld_b128(void* lds, const void* g) {
  asm volatile("global_load_async_to_lds_b128 %0, %1, off"
               :: "v"(lds_lo32(lds)), "v"((uint64_t)(uintptr_t)g) : "memory");
}
static __device__ __forceinline__ void async_wait0() {
  asm volatile("s_wait_asynccnt 0" ::: "memory");
}

// ---------------------------------------------------------------------------
// WMMA bf16 fragment helpers (layouts per CDNA5 ISA §7.12.2)
// ---------------------------------------------------------------------------
static __device__ __forceinline__ v8f wmma_bf16(v16bf a, v16bf b, v8f c) {
  return __builtin_amdgcn_wmma_f32_16x16x32_bf16(false, a, false, b, (short)0, c,
                                                 false, false);
}

// A: 16x32 bf16 tile, row-major stride lda. Per-lane data = two contiguous
// 8-element runs: k in [kh, kh+8) and [16+kh, 16+kh+8), kh = (lane>=16)*8.
static __device__ __forceinline__ v16bf load_a_frag(const bf16* A, int lda) {
  const int lane = threadIdx.x & 31;
  const int m = lane & 15;
  const int kh = (lane >> 4) << 3;
  const v8bf lo = *(const v8bf*)(A + m * lda + kh);
  const v8bf hi = *(const v8bf*)(A + m * lda + 16 + kh);
  return __builtin_shufflevector(lo, hi, 0, 1, 2, 3, 4, 5, 6, 7, 8, 9, 10, 11,
                                 12, 13, 14, 15);
}

// prod-A fragment: elementwise ui[k] * uj[m][k] (einsum('bid,bjd,dh')).
static __device__ __forceinline__ v16bf build_prod_afrag(const bf16* ui,
                                                         const bf16* uj) {
  const int lane = threadIdx.x & 31;
  const int m = lane & 15;
  const int kh = (lane >> 4) << 3;
  const v8bf a0 = *(const v8bf*)(ui + kh);
  const v8bf a1 = *(const v8bf*)(ui + 16 + kh);
  const v8bf b0 = *(const v8bf*)(uj + m * 32 + kh);
  const v8bf b1 = *(const v8bf*)(uj + m * 32 + 16 + kh);
  const v8bf p0 = a0 * b0;
  const v8bf p1 = a1 * b1;
  return __builtin_shufflevector(p0, p1, 0, 1, 2, 3, 4, 5, 6, 7, 8, 9, 10, 11,
                                 12, 13, 14, 15);
}

// B fragment from TRANSPOSED storage Bt[n][k] (k contiguous): per-lane data is
// two contiguous 8-element runs: k in [kh, kh+8) and [16+kh, 16+kh+8), kh = (lane>=16)*8.
static __device__ __forceinline__ v16bf load_bt_frag(const bf16* Bt, int ldk) {
  const int lane = threadIdx.x & 31;
  const int n = lane & 15;
  const int kh = (lane >> 4) << 3;
  const v8bf lo = *(const v8bf*)(Bt + n * ldk + kh);
  const v8bf hi = *(const v8bf*)(Bt + n * ldk + 16 + kh);
  return __builtin_shufflevector(lo, hi, 0, 1, 2, 3, 4, 5, 6, 7, 8, 9, 10, 11,
                                 12, 13, 14, 15);
}

// ---------------------------------------------------------------------------
// Generic bf16 WMMA GEMM, B transposed ([N][Kd]); double-buffered async->LDS.
// block = 16 rows x 128 cols, 8 waves (one 16-col n-tile each)
// ---------------------------------------------------------------------------
__global__ __launch_bounds__(256) void gemm_bf16(
    const bf16* __restrict__ A, int lda, long strideA,
    const bf16* __restrict__ Bt, int ldb, long strideB,
    float* __restrict__ Cf, int ldc, long strideCf,
    bf16* __restrict__ Cb, int ldcb, long strideCb, int Kd) {
  __shared__ bf16 sA[2][16 * 32];
  __shared__ bf16 sBt[2][128 * 32];
  const int bz = blockIdx.z;
  const bf16* Ab = A + (long)bz * strideA;
  const bf16* Bb = Bt + (long)bz * strideB;
  const int row0 = blockIdx.y * 16;
  const int n0 = blockIdx.x * 128;
  const int tid = threadIdx.x, w = tid >> 5, lane = tid & 31;

  auto issue = [&](int d0, int bi) {
    if (tid < 64) {  // A chunk: 16 rows x 64B
      const int r = tid >> 2, c8 = (tid & 3) << 3;
      async_ld_b128(&sA[bi][r * 32 + c8], Ab + (long)(row0 + r) * lda + d0 + c8);
    }
#pragma unroll
    for (int q = 0; q < 2; ++q) {  // Bt chunk: 128 rows x 64B
      const int f = tid + q * 256, r = f >> 2, c8 = (f & 3) << 3;
      async_ld_b128(&sBt[bi][r * 32 + c8], Bb + (long)(n0 + r) * ldb + d0 + c8);
    }
  };

  v8f acc = {0.f, 0.f, 0.f, 0.f, 0.f, 0.f, 0.f, 0.f};
  const int nch = Kd >> 5;
  issue(0, 0);
  for (int i = 0; i < nch; ++i) {
    async_wait0();       // chunk i landed (own wave's loads)
    __syncthreads();     // all waves' chunk-i loads landed; buf (i+1)&1 free
    if (i + 1 < nch) issue((i + 1) << 5, (i + 1) & 1);
    const int bi = i & 1;
    v16bf a = load_a_frag(sA[bi], 32);
    v16bf b = load_bt_frag(&sBt[bi][(w * 16) * 32], 32);
    acc = wmma_bf16(a, b, acc);
  }

  const int nloc = lane & 15, rbase = (lane >> 4) << 3;
#pragma unroll
  for (int v = 0; v < 8; ++v) {
    const long r = row0 + rbase + v;
    const int cc = n0 + w * 16 + nloc;
    const float x = acc[v];
    if (Cf) Cf[(long)bz * strideCf + r * ldc + cc] = x;
    if (Cb) Cb[(long)bz * strideCb + r * ldcb + cc] = (bf16)x;
  }
}

// ---------------------------------------------------------------------------
// Fused pair kernel: prod(+left+right+dist) -> relu -> @W_h -> relu -> @W_out
// grid (K/16, K/16, B); block 256 (8 waves); wave w owns i_local {2w, 2w+1}
// ---------------------------------------------------------------------------
__global__ __launch_bounds__(256) void k_pair(
    const bf16* __restrict__ cat,       // [B*K][2*D], cols 0..D-1 = u (bf16)
    const float* __restrict__ LR,       // [B*K][256]: left | right
    const int* __restrict__ bucket,     // [B*K*K]
    const float* __restrict__ dist_tab, // [NB*H] (biases folded)
    const bf16* __restrict__ WprodT,    // [H][D] transposed
    const bf16* __restrict__ WhT,       // [H][H] transposed
    const float* __restrict__ b_h, const float* __restrict__ W_out,
    const float* __restrict__ b_out, float* __restrict__ scores) {
  union ShU {
    struct { bf16 ui[2][16 * 32]; bf16 uj[2][16 * 32]; bf16 wp[2][128 * 32]; } lp;
    struct { bf16 stage[8][16 * 128]; bf16 wh[128 * 128]; } ep;
  };
  __shared__ ShU sh;

  const int b = blockIdx.z, it = blockIdx.y, jt = blockIdx.x;
  const int i0 = it * 16, j0 = jt * 16;
  const int tid = threadIdx.x, w = tid >> 5, lane = tid & 31;
  const int nloc = lane & 15, half8 = (lane >> 4) << 3;

  auto issue = [&](int d0, int bi) {
    if (tid < 128) {  // ui / uj chunks: 16 rows x 64B each
      const int sel = tid >> 6, t = tid & 63;
      const int r = t >> 2, c8 = (t & 3) << 3;
      const int row = (sel ? j0 : i0) + r;
      bf16* dst = (sel ? sh.lp.uj[bi] : sh.lp.ui[bi]) + r * 32 + c8;
      async_ld_b128(dst, cat + (long)(b * Kc + row) * (2 * Dc) + d0 + c8);
    }
#pragma unroll
    for (int q = 0; q < 2; ++q) {  // WprodT chunk: 128 rows x 64B
      const int f = tid + q * 256, r = f >> 2, c8 = (f & 3) << 3;
      async_ld_b128(&sh.lp.wp[bi][r * 32 + c8], WprodT + (long)r * Dc + d0 + c8);
    }
  };

  const v8f vz = {0.f, 0.f, 0.f, 0.f, 0.f, 0.f, 0.f, 0.f};
  v8f c[2][8];
#pragma unroll
  for (int mt = 0; mt < 2; ++mt)
#pragma unroll
    for (int nt = 0; nt < 8; ++nt) c[mt][nt] = vz;

  const int nch = Dc >> 5;
  issue(0, 0);
  for (int i = 0; i < nch; ++i) {
    async_wait0();
    __syncthreads();
    if (i + 1 < nch) issue((i + 1) << 5, (i + 1) & 1);
    const int bi = i & 1;
    v16bf a0 = build_prod_afrag(&sh.lp.ui[bi][(2 * w) * 32], sh.lp.uj[bi]);
    v16bf a1 = build_prod_afrag(&sh.lp.ui[bi][(2 * w + 1) * 32], sh.lp.uj[bi]);
#pragma unroll
    for (int nt = 0; nt < 8; ++nt) {
      v16bf bf_ = load_bt_frag(&sh.lp.wp[bi][(nt * 16) * 32], 32);
      c[0][nt] = wmma_bf16(a0, bf_, c[0][nt]);
      c[1][nt] = wmma_bf16(a1, bf_, c[1][nt]);
    }
  }

  // Epilogue: bring transposed W_h into LDS (re-uses loop LDS via union)
  __syncthreads();
#pragma unroll
  for (int q = 0; q < 8; ++q) {
    const int f = tid + q * 256, r = f >> 4, c8 = (f & 15) << 3;
    async_ld_b128(&sh.ep.wh[r * 128 + c8], WhT + (long)r * Hc + c8);
  }
  async_wait0();
  __syncthreads();

  const float bo = b_out[0];
#pragma unroll
  for (int mt = 0; mt < 2; ++mt) {
    __syncthreads();  // protect per-wave stage reuse across mt iterations
    const int gi = i0 + 2 * w + mt;
    bf16* stg = &sh.ep.stage[w][0];
#pragma unroll
    for (int nt = 0; nt < 8; ++nt) {
      const int h = nt * 16 + nloc;
      const float l = LR[(long)(b * Kc + gi) * 256 + h];
#pragma unroll
      for (int v = 0; v < 8; ++v) {
        const int jl = half8 + v, gj = j0 + jl;
        const float r = LR[(long)(b * Kc + gj) * 256 + 128 + h];
        const int bk = bucket[(long)(b * Kc + gi) * Kc + gj];
        const float dt = dist_tab[bk * Hc + h];
        const float h1 = c[mt][nt][v] + l + r + dt;
        stg[jl * 128 + h] = (bf16)fmaxf(h1, 0.f);
      }
    }
    __syncthreads();
    v8f c2[8];
#pragma unroll
    for (int nt = 0; nt < 8; ++nt) c2[nt] = vz;
#pragma unroll
    for (int kc = 0; kc < 4; ++kc) {
      v16bf a = load_a_frag(stg + kc * 32, 128);
#pragma unroll
      for (int nt = 0; nt < 8; ++nt) {
        v16bf bf_ = load_bt_frag(&sh.ep.wh[(nt * 16) * 128 + kc * 32], 128);
        c2[nt] = wmma_bf16(a, bf_, c2[nt]);
      }
    }
    // score = relu(h2 + b_h) . W_out  (reduce over h = lanes x n-tiles)
    float accv[8];
#pragma unroll
    for (int v = 0; v < 8; ++v) accv[v] = 0.f;
#pragma unroll
    for (int nt = 0; nt < 8; ++nt) {
      const int h = nt * 16 + nloc;
      const float wo = W_out[h], bh = b_h[h];
#pragma unroll
      for (int v = 0; v < 8; ++v) accv[v] += fmaxf(c2[nt][v] + bh, 0.f) * wo;
    }
#pragma unroll
    for (int m = 1; m < 16; m <<= 1)
#pragma unroll
      for (int v = 0; v < 8; ++v) accv[v] += __shfl_xor(accv[v], m, 32);
    if (nloc < 8) {
      const int jl = half8 + nloc;
      float sc = 0.f;
#pragma unroll
      for (int v = 0; v < 8; ++v) sc = (v == nloc) ? accv[v] : sc;
      scores[(long)(b * Kc + gi) * Kc + j0 + jl] = sc + bo;
    }
  }
}

// ---------------------------------------------------------------------------
// Small support kernels
// ---------------------------------------------------------------------------
// Transpose-convert: src f32 [R][C] -> dst bf16 [C][R]
__global__ void k_cvt_T(const float* __restrict__ s, bf16* __restrict__ d,
                        long R, long C) {
  const long i = (long)blockIdx.x * blockDim.x + threadIdx.x;
  if (i < R * C) {
    const long r = i / C, c = i % C;
    d[c * R + r] = (bf16)s[i];
  }
}

// W_left|W_right -> transposed bf16 [256][D]
__global__ void k_cvt_lrT(const float* __restrict__ wl,
                          const float* __restrict__ wr,
                          bf16* __restrict__ wlrT) {
  const long i = (long)blockIdx.x * blockDim.x + threadIdx.x;
  if (i < (long)Dc * Hc) {
    const long dd = i >> 7, h = i & 127;
    wlrT[h * Dc + dd] = (bf16)wl[i];
    wlrT[(Hc + h) * Dc + dd] = (bf16)wr[i];
  }
}

// u (f32 [B*K][D]) -> cat left half bf16 + u_T bf16 [B][D][K]
__global__ void k_cvt_cat(const float* __restrict__ u, bf16* __restrict__ cat,
                          bf16* __restrict__ uT) {
  const long i = (long)blockIdx.x * blockDim.x + threadIdx.x;
  if (i < (long)Bc * Kc * Dc) {
    const long row = i >> 10, d = i & 1023;
    const long b = row >> 8, k = row & 255;
    const bf16 v = (bf16)u[i];
    cat[row * (2 * Dc) + d] = v;
    uT[(b * Dc + d) * Kc + k] = v;
  }
}

__global__ void k_dist_tab(const float* __restrict__ de,
                           const float* __restrict__ wd,
                           const float* __restrict__ bd,
                           const float* __restrict__ bl,
                           const float* __restrict__ br,
                           const float* __restrict__ bp,
                           float* __restrict__ tab) {
  const int i = blockIdx.x * blockDim.x + threadIdx.x;
  if (i < NBc * Hc) {
    const int nb = i / Hc, h = i % Hc;
    float acc = bd[h] + bl[h] + br[h] + bp[h];
    for (int e = 0; e < DEc; ++e) acc += de[nb * DEc + e] * wd[e * Hc + h];
    tab[i] = acc;
  }
}

__global__ void k_bucket(const int* __restrict__ sb, const int* __restrict__ se,
                         int* __restrict__ bkt) {
  const long idx = (long)blockIdx.x * blockDim.x + threadIdx.x;
  if (idx < (long)Bc * Kc * Kc) {
    const int j = idx % Kc, i = (idx / Kc) % Kc, b = idx / (Kc * Kc);
    int g1 = sb[b * Kc + j] - se[b * Kc + i];
    int g2 = sb[b * Kc + i] - se[b * Kc + j];
    int g = g1 > g2 ? g1 : g2;
    if (g < 0) g = 0;
    int bk;
    if (g < 5) bk = g;
    else {
      bk = (31 - __clz(g)) + 3;
      if (bk > NBc - 1) bk = NBc - 1;
    }
    bkt[idx] = bk;
  }
}

__global__ __launch_bounds__(256) void k_softmax(const float* __restrict__ scores,
                                                 const float* __restrict__ mask,
                                                 bf16* __restrict__ probs) {
  const int b = blockIdx.y, i = blockIdx.x, j = threadIdx.x;
  const long o = (long)(b * Kc + i) * Kc + j;
  const float s = scores[o] - (1.0f - mask[o]) * 1e23f;
  __shared__ float red[256];
  red[j] = s;
  __syncthreads();
  for (int st = 128; st > 0; st >>= 1) {
    if (j < st) red[j] = fmaxf(red[j], red[j + st]);
    __syncthreads();
  }
  const float mx = red[0];
  __syncthreads();
  const float e = __expf(s - mx);
  red[j] = e;
  __syncthreads();
  for (int st = 128; st > 0; st >>= 1) {
    if (j < st) red[j] += red[j + st];
    __syncthreads();
  }
  probs[o] = (bf16)(e / red[0]);
}

__global__ void k_gate_epi(const float* __restrict__ glog,
                           const float* __restrict__ b_gate,
                           const float* __restrict__ u_cur,
                           const float* __restrict__ ctxt,
                           float* __restrict__ u_next, bf16* __restrict__ cat,
                           bf16* __restrict__ uT) {
  const long i = (long)blockIdx.x * blockDim.x + threadIdx.x;
  if (i < (long)Bc * Kc * Dc) {
    const long d = i & 1023, row = i >> 10;
    const long b = row >> 8, k = row & 255;
    const float g = 1.0f / (1.0f + __expf(-(glog[i] + b_gate[d])));
    const float un = g * u_cur[i] + (1.0f - g) * ctxt[i];
    u_next[i] = un;
    const bf16 v = (bf16)un;
    cat[row * (2 * Dc) + d] = v;
    uT[(b * Dc + d) * Kc + k] = v;
  }
}

__global__ void k_copyf(const float* __restrict__ s, float* __restrict__ d,
                        long n) {
  const long i = (long)blockIdx.x * blockDim.x + threadIdx.x;
  if (i < n) d[i] = s[i];
}

__global__ void k_scatter(const float* __restrict__ upd,
                          const int* __restrict__ pidx,
                          const int* __restrict__ slen,
                          float* __restrict__ out_all) {
  const int k = blockIdx.x, b = blockIdx.y;
  if (k < slen[b]) {
    const int pi = pidx[b * Kc + k];
    for (int d = threadIdx.x; d < Dc; d += 256)
      out_all[((long)b * Nc + pi) * Dc + d] = upd[((long)b * Kc + k) * Dc + d];
  }
}

// ---------------------------------------------------------------------------
// Host orchestration
// ---------------------------------------------------------------------------
extern "C" void kernel_launch(void* const* d_in, const int* in_sizes, int n_in,
                              void* d_out, int out_size, void* d_ws,
                              size_t ws_size, hipStream_t stream) {
  (void)in_sizes; (void)n_in; (void)out_size; (void)ws_size;
  const float* span_vecs   = (const float*)d_in[0];
  const float* square_mask = (const float*)d_in[1];
  const float* all_span    = (const float*)d_in[2];
  const int*   span_begin  = (const int*)d_in[3];
  const int*   span_end    = (const int*)d_in[4];
  const int*   prune_idx   = (const int*)d_in[5];
  const int*   span_len    = (const int*)d_in[6];
  const float* W_left  = (const float*)d_in[7];
  const float* b_left  = (const float*)d_in[8];
  const float* W_right = (const float*)d_in[9];
  const float* b_right = (const float*)d_in[10];
  const float* W_prod  = (const float*)d_in[11];
  const float* b_prod  = (const float*)d_in[12];
  const float* dist_emb= (const float*)d_in[13];
  const float* W_dist  = (const float*)d_in[14];
  const float* b_dist  = (const float*)d_in[15];
  const float* W_h     = (const float*)d_in[16];
  const float* b_h     = (const float*)d_in[17];
  const float* W_out   = (const float*)d_in[18];
  const float* b_out   = (const float*)d_in[19];
  const float* W_gate  = (const float*)d_in[20];
  const float* b_gate  = (const float*)d_in[21];

  char* p = (char*)d_ws;
  auto alloc = [&](size_t bytes) -> void* {
    void* r = (void*)p;
    p += (bytes + 255) & ~(size_t)255;
    return r;
  };
  float* dist_tab = (float*)alloc((size_t)NBc * Hc * 4);
  int*   bucket   = (int*)alloc((size_t)Bc * Kc * Kc * 4);
  bf16*  WlrT     = (bf16*)alloc((size_t)2 * Hc * Dc * 2);   // [256][1024]
  bf16*  WprodT   = (bf16*)alloc((size_t)Hc * Dc * 2);       // [128][1024]
  bf16*  WhT      = (bf16*)alloc((size_t)Hc * Hc * 2);       // [128][128]
  bf16*  WgateT   = (bf16*)alloc((size_t)Dc * 2 * Dc * 2);   // [1024][2048]
  bf16*  cat      = (bf16*)alloc((size_t)Bc * Kc * 2 * Dc * 2);
  bf16*  uT       = (bf16*)alloc((size_t)Bc * Dc * Kc * 2);  // [B][D][K]
  float* LR       = (float*)alloc((size_t)Bc * Kc * 256 * 4);
  float* scores   = (float*)alloc((size_t)Bc * Kc * Kc * 4);
  bf16*  probs_bf = (bf16*)alloc((size_t)Bc * Kc * Kc * 2);
  float* ctxt     = (float*)alloc((size_t)Bc * Kc * Dc * 4);
  float* glog     = (float*)alloc((size_t)Bc * Kc * Dc * 4);
  float* upd      = (float*)alloc((size_t)Bc * Kc * Dc * 4);

  const long nWlr = (long)Dc * Hc;
  k_cvt_lrT<<<(nWlr + 255) / 256, 256, 0, stream>>>(W_left, W_right, WlrT);
  k_cvt_T<<<(nWlr + 255) / 256, 256, 0, stream>>>(W_prod, WprodT, Dc, Hc);
  k_cvt_T<<<((long)Hc * Hc + 255) / 256, 256, 0, stream>>>(W_h, WhT, Hc, Hc);
  k_cvt_T<<<((long)2 * Dc * Dc + 255) / 256, 256, 0, stream>>>(W_gate, WgateT,
                                                               2 * Dc, Dc);
  k_dist_tab<<<(NBc * Hc + 255) / 256, 256, 0, stream>>>(
      dist_emb, W_dist, b_dist, b_left, b_right, b_prod, dist_tab);
  k_bucket<<<((long)Bc * Kc * Kc + 255) / 256, 256, 0, stream>>>(span_begin,
                                                                 span_end, bucket);
  const long nU = (long)Bc * Kc * Dc;
  k_cvt_cat<<<(nU + 255) / 256, 256, 0, stream>>>(span_vecs, cat, uT);

  const float* u_cur = span_vecs;
  for (int t = 0; t < 2; ++t) {
    // left|right projection: [512,1024] x [1024,256] (B transposed)
    gemm_bf16<<<dim3(2, 32, 1), 256, 0, stream>>>(
        cat, 2 * Dc, 0, WlrT, Dc, 0, LR, 256, 0, nullptr, 0, 0, Dc);
    // fused pair scores
    k_pair<<<dim3(Kc / 16, Kc / 16, Bc), 256, 0, stream>>>(
        cat, LR, bucket, dist_tab, WprodT, WhT, b_h, W_out, b_out, scores);
    // softmax over j
    k_softmax<<<dim3(Kc, Bc), 256, 0, stream>>>(scores, square_mask, probs_bf);
    // ctxt = probs @ update  (batched): [256,256] x [256,1024] per b
    gemm_bf16<<<dim3(Dc / 128, Kc / 16, Bc), 256, 0, stream>>>(
        probs_bf, Kc, (long)Kc * Kc, uT, Kc, (long)Dc * Kc,
        ctxt, Dc, (long)Kc * Dc, cat + Dc, 2 * Dc, (long)Kc * 2 * Dc, Kc);
    // gate logits: [512,2048] x [2048,1024]
    gemm_bf16<<<dim3(Dc / 128, (Bc * Kc) / 16, 1), 256, 0, stream>>>(
        cat, 2 * Dc, 0, WgateT, 2 * Dc, 0, glog, Dc, 0, nullptr, 0, 0, 2 * Dc);
    // sigmoid blend -> new update (f32 + bf16 into cat and uT)
    k_gate_epi<<<(nU + 255) / 256, 256, 0, stream>>>(glog, b_gate, u_cur, ctxt,
                                                     upd, cat, uT);
    u_cur = upd;
  }

  // outputs: [B,N,D] scatter-updated all spans, then [B,K,D] update
  float* out_all = (float*)d_out;
  float* out_upd = out_all + (long)Bc * Nc * Dc;
  k_copyf<<<(((long)Bc * Nc * Dc) + 255) / 256, 256, 0, stream>>>(
      all_span, out_all, (long)Bc * Nc * Dc);
  k_copyf<<<(nU + 255) / 256, 256, 0, stream>>>(upd, out_upd, nU);
  k_scatter<<<dim3(Kc, Bc), 256, 0, stream>>>(upd, prune_idx, span_len, out_all);
}
